// DifferentiablePhysicsUpdate_40492951666855
// MI455X (gfx1250) — matrix operand order repaired and checked by reading, not compiled, not run on hardware
//
#include <hip/hip_runtime.h>
#include <math.h>

// ---------------------------------------------------------------------------
// DifferentiablePhysicsUpdate for MI455X (gfx1250, wave32, WMMA)
//   out = concat( h + dh  (B*N floats),  flows (B*E floats) )
// ---------------------------------------------------------------------------

typedef __attribute__((ext_vector_type(16))) __bf16 v16bf;
typedef __attribute__((ext_vector_type(8)))  float  v8f;

static constexpr int HDIM = 128;   // hidden
static constexpr int KDIM = 64;    // hidden/2
static constexpr int BATCH = 2;
static constexpr float DT_F = 300.0f;
static constexpr float MAXFLOW_F = 10.0f;

__device__ __forceinline__ __bf16 f2bf(float f) {
  return static_cast<__bf16>(f);   // gfx1250 has native f32->bf16 cvt
}

// ---------------------------------------------------------------------------
// Kernel 1: per-edge MLP (WMMA bf16) + flow + scatter-add of net flow
// One wave handles a tile of 16 edges for one batch.
// ---------------------------------------------------------------------------
__global__ __launch_bounds__(256)
void edge_mlp_flow_kernel(const float* __restrict__ h,
                          const float* __restrict__ z,
                          const int*   __restrict__ edge_index,  // [2*E]: src then dst
                          const float* __restrict__ W1,          // [128*64] row-major
                          const float* __restrict__ b1,          // [64]
                          const float* __restrict__ W2,          // [64]
                          const float* __restrict__ b2,          // [1]
                          float* __restrict__ flows_out,         // [B*E]
                          float* __restrict__ net,               // [B*N], pre-zeroed
                          int N, int E)
{
  __shared__ __bf16 w1t[KDIM * HDIM];  // transposed: w1t[col][k], 16 KB
  __shared__ float  b1s[KDIM];
  __shared__ float  w2s[KDIM];
  __shared__ float  b2s;

  const int tid = threadIdx.x;

  // Cooperative: convert W1 (f32, [k][c]) -> bf16 transposed [c][k] in LDS.
  for (int i = tid; i < HDIM * KDIM; i += 256) {
    int k = i / KDIM, c = i - k * KDIM;
    w1t[c * HDIM + k] = f2bf(W1[i]);
  }
  if (tid < KDIM) { b1s[tid] = b1[tid]; w2s[tid] = W2[tid]; }
  if (tid == 0)   { b2s = b2[0]; }
  __syncthreads();

  const int lane = tid & 31;
  const int col  = lane & 15;   // column (B/C/D) / row (A) index
  const int half = lane >> 4;   // which K half this lane holds

  // Hoist all 16 B-matrix fragments of W1 (bf16) into registers.
  // B 32x16 in contiguous K order (same as A): lanes 0-15 hold K=0..15, lanes 16-31 hold K=16..31,
  // contiguous per lane -> one aligned 32-byte LDS read each.
  v16bf bfr[4][4];
  #pragma unroll
  for (int kk = 0; kk < 4; ++kk)
    #pragma unroll
    for (int n = 0; n < 4; ++n)
      bfr[kk][n] = *(const v16bf*)&w1t[(n * 16 + col) * HDIM + kk * 32 + half * 16];

  const int tilesPerBatch = (E + 15) >> 4;
  const int T = BATCH * tilesPerBatch;
  const int wave   = (blockIdx.x * blockDim.x + tid) >> 5;
  const int nWaves = (gridDim.x * blockDim.x) >> 5;

  for (int t = wave; t < T; t += nWaves) {   // wave-uniform loop
    const int b   = t / tilesPerBatch;
    const int e0  = (t - b * tilesPerBatch) << 4;
    const int myE = e0 + col;                     // this lane's edge/row
    const int eg  = (myE < E) ? myE : (E - 1);    // clamp (keeps EXEC all-ones)
    const int s   = edge_index[eg];               // src node of my row
    const float* zr = z + ((size_t)b * N + s) * (size_t)HDIM;

    v8f acc[4];
    #pragma unroll
    for (int n = 0; n < 4; ++n) {
      v8f zv = {0.f,0.f,0.f,0.f,0.f,0.f,0.f,0.f};
      acc[n] = zv;
    }

    // K loop: 4 steps of K=32. A fragment (contiguous, matching B):
    //   elems 0..15 = K = kk*32 + half*16 + 0..15
    #pragma unroll
    for (int kk = 0; kk < 4; ++kk) {
      const int c0 = kk * 32 + half * 16;
      const float4 f0 = *(const float4*)(zr + c0);
      const float4 f1 = *(const float4*)(zr + c0 + 4);
      const float4 f2 = *(const float4*)(zr + c0 + 8);
      const float4 f3 = *(const float4*)(zr + c0 + 12);
      v16bf a;
      a[0]=f2bf(f0.x);  a[1]=f2bf(f0.y);  a[2]=f2bf(f0.z);  a[3]=f2bf(f0.w);
      a[4]=f2bf(f1.x);  a[5]=f2bf(f1.y);  a[6]=f2bf(f1.z);  a[7]=f2bf(f1.w);
      a[8]=f2bf(f2.x);  a[9]=f2bf(f2.y);  a[10]=f2bf(f2.z); a[11]=f2bf(f2.w);
      a[12]=f2bf(f3.x); a[13]=f2bf(f3.y); a[14]=f2bf(f3.z); a[15]=f2bf(f3.w);

      #pragma unroll
      for (int n = 0; n < 4; ++n)
        acc[n] = __builtin_amdgcn_wmma_f32_16x16x32_bf16(
            /*neg_a=*/false, a, /*neg_b=*/false, bfr[kk][n],
            /*c_mod=*/(short)0, acc[n], /*reuse_a=*/false, /*reuse_b=*/false);
    }

    // Epilogue: bias + exact gelu, dot with W2 column weights, reduce columns.
    // C layout: VGPR r holds rows r (lanes 0-15) / r+8 (lanes 16-31), col = lane%16.
    float dotp[8];
    #pragma unroll
    for (int r = 0; r < 8; ++r) dotp[r] = 0.f;
    #pragma unroll
    for (int n = 0; n < 4; ++n) {
      const float bias = b1s[n * 16 + col];
      const float w2v  = w2s[n * 16 + col];
      #pragma unroll
      for (int r = 0; r < 8; ++r) {
        const float x = acc[n][r] + bias;
        const float g = 0.5f * x * (1.0f + erff(x * 0.70710678118654752f));
        dotp[r] = fmaf(g, w2v, dotp[r]);
      }
    }
    #pragma unroll
    for (int r = 0; r < 8; ++r) {
      #pragma unroll
      for (int m = 8; m >= 1; m >>= 1)
        dotp[r] += __shfl_xor(dotp[r], m, 32);
    }

    // Row m lives in dotp[m&7] of lane (m>>3)*16. Give lane `col` row `col`.
    float mydot = 0.f;
    #pragma unroll
    for (int m = 0; m < 16; ++m) {
      const float v = __shfl(dotp[m & 7], (m >> 3) << 4, 32);
      if (col == m) mydot = v;
    }

    // Per-edge tail on lanes 0..15 (divergence only after last WMMA).
    if (half == 0 && myE < E) {
      const int sE = edge_index[myE];
      const int dE = edge_index[E + myE];
      const float dh  = h[(size_t)b * N + sE] - h[(size_t)b * N + dE];
      const float sig = 1.0f / (1.0f + expf(-(mydot + b2s)));
      const float sg  = (dh > 0.f) ? 1.f : ((dh < 0.f) ? -1.f : 0.f);
      float fl = sig * sg * sqrtf(fabsf(dh) + 1e-6f);
      fl = fminf(fmaxf(fl, -MAXFLOW_F), MAXFLOW_F);
      flows_out[(size_t)b * E + myE] = fl;
      // inflow - outflow collapses to: net[dst] += flow, net[src] -= flow
      atomicAdd(&net[(size_t)b * N + dE],  fl);
      atomicAdd(&net[(size_t)b * N + sE], -fl);
    }
  }
}

// ---------------------------------------------------------------------------
// Kernel 2: per-node areas (softplus(z·Wa+ba)) and final h update.
// One wave per node: float4 loads + full-wave shuffle reduction.
// ---------------------------------------------------------------------------
__global__ __launch_bounds__(256)
void node_update_kernel(const float* __restrict__ h,
                        const float* __restrict__ z,
                        const float* __restrict__ rainfall,
                        const float* __restrict__ Wa,   // [128]
                        const float* __restrict__ ba,   // [1]
                        const float* __restrict__ net,  // [B*N]
                        float* __restrict__ h_out,      // [B*N]
                        int N)
{
  const int wave = (blockIdx.x * blockDim.x + threadIdx.x) >> 5;  // flat b*N+n
  const int lane = threadIdx.x & 31;
  if (wave >= BATCH * N) return;   // wave-uniform

  const float* zr = z + (size_t)wave * HDIM;
  const float4 zv = *(const float4*)(zr + lane * 4);
  const float4 wv = *(const float4*)(Wa + lane * 4);
  float d = zv.x * wv.x + zv.y * wv.y + zv.z * wv.z + zv.w * wv.w;
  #pragma unroll
  for (int m = 16; m >= 1; m >>= 1) d += __shfl_xor(d, m, 32);

  if (lane == 0) {
    const float x = d + ba[0];
    // numerically-stable softplus
    const float area = fmaxf(x, 0.f) + log1pf(expf(-fabsf(x)));
    const float nf = net[wave] + rainfall[wave];
    float dh = DT_F * nf / (area + 1e-6f);
    dh = fminf(fmaxf(dh, -1.f), 1.f);
    h_out[wave] = h[wave] + dh;
  }
}

// ---------------------------------------------------------------------------
extern "C" void kernel_launch(void* const* d_in, const int* in_sizes, int n_in,
                              void* d_out, int out_size, void* d_ws, size_t ws_size,
                              hipStream_t stream) {
  const float* h    = (const float*)d_in[0];   // [B*N]
  const float* z    = (const float*)d_in[1];   // [B*N*128]
  const int*   ei   = (const int*)  d_in[2];   // [2*E]
  /* d_in[3] = edge_type (unused by reference math) */
  const float* rain = (const float*)d_in[4];   // [B*N]
  const float* W1   = (const float*)d_in[5];   // [128*64]
  const float* b1   = (const float*)d_in[6];   // [64]
  const float* W2   = (const float*)d_in[7];   // [64]
  const float* b2   = (const float*)d_in[8];   // [1]
  const float* Wa   = (const float*)d_in[9];   // [128]
  const float* ba   = (const float*)d_in[10];  // [1]

  const int N = in_sizes[0] / BATCH;
  const int E = in_sizes[3];

  float* out_h     = (float*)d_out;                    // [B*N]
  float* out_flows = out_h + (size_t)BATCH * N;        // [B*E]
  float* net       = (float*)d_ws;                     // [B*N] scratch

  hipMemsetAsync(net, 0, sizeof(float) * (size_t)BATCH * N, stream);

  const int tiles   = BATCH * ((E + 15) / 16);
  int blocksE = (tiles + 7) / 8;           // 8 waves (256 thr) per block
  if (blocksE > 4096) blocksE = 4096;      // grid-stride amortizes LDS/B-frag setup
  edge_mlp_flow_kernel<<<blocksE, 256, 0, stream>>>(
      h, z, ei, W1, b1, W2, b2, out_flows, net, N, E);

  const int wavesN  = BATCH * N;
  const int blocksN = (wavesN + 7) / 8;
  node_update_kernel<<<blocksN, 256, 0, stream>>>(
      h, z, rain, Wa, ba, net, out_h, N);
}
